// HunyuanTopKGate_82764019794349
// MI455X (gfx1250) — matrix operand order repaired and checked by reading, not compiled, not run on hardware
//
#include <hip/hip_runtime.h>
#include <hip/hip_bf16.h>

// ---------------- static problem config (matches reference) ----------------
#define HIDDEN    4096
#define NEXP      64
#define TOPK      8
#define NGRP      8
#define TKG       4
#define TTOK      16384          // B*S = 4*4096
#define CAPACITY  2048           // int(max(8, 8*16384//64) * 1.0)
#define TPB       128            // tokens per block in gate kernel
#define NBLK      (TTOK / TPB)   // 128
#define KITERS    (HIDDEN / 32)  // 128 wmma K-steps
#define TE        (TTOK * NEXP)  // 1048576 dense elements per [T,E] output

typedef __attribute__((ext_vector_type(16))) __bf16 v16bf;
typedef __attribute__((ext_vector_type(8)))  float  v8f;

__device__ __forceinline__ unsigned ballot32(bool p) {
#if __has_builtin(__builtin_amdgcn_ballot_w32)
  return __builtin_amdgcn_ballot_w32(p);
#else
  return (unsigned)__ballot(p);
#endif
}

// ---------------------------------------------------------------------------
// Kernel 1: zero dense outputs (d_out poisoned by harness) + valid counter
// ---------------------------------------------------------------------------
__global__ void hy_zero_kernel(float* __restrict__ out, int n,
                               int* __restrict__ valid_count) {
  int i = blockIdx.x * blockDim.x + threadIdx.x;
  int stride = gridDim.x * blockDim.x;
  for (int j = i; j < n; j += stride) out[j] = 0.0f;
  if (i == 0) *valid_count = 0;
}

// ---------------------------------------------------------------------------
// Kernel 2: convert W (fp32 [64,4096] row-major) into bf16 B-fragment layout
//   Wb elem index f = ((nt*KITERS + kit)*32 + lane)*16 + i
//   B 32x16 bf16 layout: lane 0-15 -> N=lane, K = kb + 0..7, 16..23 (2/VGPR)
//                        lane 16-31 -> N=lane-16, K = kb + 8..15, 24..31
// ---------------------------------------------------------------------------
__global__ void hy_prep_w_kernel(const float* __restrict__ W,
                                 __bf16* __restrict__ Wb) {
  int f = blockIdx.x * blockDim.x + threadIdx.x;   // 0 .. 4*128*32*16-1
  int i    = f & 15;
  int lane = (f >> 4) & 31;
  int kit  = (f >> 9) & (KITERS - 1);
  int nt   = f >> 16;
  int n = nt * 16 + (lane & 15);
  int k = kit * 32 + ((lane >> 4) << 3) + (i & 7) + ((i >> 3) << 4);
  Wb[f] = (__bf16)W[(size_t)n * HIDDEN + k];
}

// ---------------------------------------------------------------------------
// Kernel 3: gate GEMM (bf16 WMMA) + softmax + group-limited top-k
// Block: 256 threads = 8 waves; wave w owns tokens [blk*128 + w*16, +16)
// and all 64 experts via 4 accumulators.
// ---------------------------------------------------------------------------
__global__ __launch_bounds__(256) void hy_gate_kernel(
    const float* __restrict__ X, const __bf16* __restrict__ Wb,
    int* __restrict__ sel, float* __restrict__ rp,
    int* __restrict__ hist, float* __restrict__ gsum_blk) {
  __shared__ float lds[TPB][NEXP + 1];   // +1 pad: conflict-free row access
  __shared__ int   lhist[TOPK][NEXP];

  const int lane  = threadIdx.x & 31;
  const int wave  = threadIdx.x >> 5;
  const int blk   = blockIdx.x;
  const int khalf = lane >> 4;

  // ---------------- phase 1: logits via v_wmma_f32_16x16x32_bf16 ----------
  v8f acc0 = {}, acc1 = {}, acc2 = {}, acc3 = {};
  // A 16x32 bf16 layout: lane holds row M=(lane&15); K runs
  //   kb + khalf*8 + {0..7}  (elems 0..7)  and  kb + 16 + khalf*8 + {0..7}
  const float* xrow =
      X + (size_t)(blk * TPB + wave * 16 + (lane & 15)) * HIDDEN + khalf * 8;
  for (int kit = 0; kit < KITERS; ++kit) {
    const float* xp = xrow + kit * 32;
    float4 a0 = *(const float4*)(xp);
    float4 a1 = *(const float4*)(xp + 4);
    float4 a2 = *(const float4*)(xp + 16);
    float4 a3 = *(const float4*)(xp + 20);
    v16bf af;
    af[0]=(__bf16)a0.x;  af[1]=(__bf16)a0.y;  af[2]=(__bf16)a0.z;  af[3]=(__bf16)a0.w;
    af[4]=(__bf16)a1.x;  af[5]=(__bf16)a1.y;  af[6]=(__bf16)a1.z;  af[7]=(__bf16)a1.w;
    af[8]=(__bf16)a2.x;  af[9]=(__bf16)a2.y;  af[10]=(__bf16)a2.z; af[11]=(__bf16)a2.w;
    af[12]=(__bf16)a3.x; af[13]=(__bf16)a3.y; af[14]=(__bf16)a3.z; af[15]=(__bf16)a3.w;

    const __bf16* wp = Wb + (size_t)kit * 512 + lane * 16;
    v16bf w0 = *(const v16bf*)(wp);
    v16bf w1 = *(const v16bf*)(wp + 1 * KITERS * 512);
    v16bf w2 = *(const v16bf*)(wp + 2 * KITERS * 512);
    v16bf w3 = *(const v16bf*)(wp + 3 * KITERS * 512);

    acc0 = __builtin_amdgcn_wmma_f32_16x16x32_bf16(false, af, false, w0, (short)0, acc0, false, false);
    acc1 = __builtin_amdgcn_wmma_f32_16x16x32_bf16(false, af, false, w1, (short)0, acc1, false, false);
    acc2 = __builtin_amdgcn_wmma_f32_16x16x32_bf16(false, af, false, w2, (short)0, acc2, false, false);
    acc3 = __builtin_amdgcn_wmma_f32_16x16x32_bf16(false, af, false, w3, (short)0, acc3, false, false);
  }

  // C/D layout: VGPR r -> (M = r + 8*khalf, N = lane&15)
#pragma unroll
  for (int r = 0; r < 8; ++r) {
    int tl = wave * 16 + r + khalf * 8;
    int c  = lane & 15;
    lds[tl][c]      = acc0[r];
    lds[tl][c + 16] = acc1[r];
    lds[tl][c + 32] = acc2[r];
    lds[tl][c + 48] = acc3[r];
  }
  for (int i = threadIdx.x; i < TOPK * NEXP; i += blockDim.x)
    ((int*)lhist)[i] = 0;
  __syncthreads();

  // ---------------- phase 2: per-token softmax / group top-k --------------
  if (threadIdx.x < TPB) {
    float* row = lds[threadIdx.x];
    float mx = row[0];
    for (int e = 1; e < NEXP; ++e) mx = fmaxf(mx, row[e]);
    float s = 0.0f;
    for (int e = 0; e < NEXP; ++e) { float v = __expf(row[e] - mx); row[e] = v; s += v; }
    float inv = 1.0f / s;
    for (int e = 0; e < NEXP; ++e) row[e] *= inv;   // gates (softmax)

    // group scores = max over each group of 8
    float gm[NGRP];
    for (int g = 0; g < NGRP; ++g) {
      float m2 = row[g * 8];
      for (int j = 1; j < 8; ++j) m2 = fmaxf(m2, row[g * 8 + j]);
      gm[g] = m2;
    }
    // top-4 groups (strict > matches stable top_k tie-break)
    unsigned gmask = 0;
    for (int it = 0; it < TKG; ++it) {
      int best = 0; float bv = -1.0f;
      for (int g = 0; g < NGRP; ++g)
        if (!((gmask >> g) & 1u) && gm[g] > bv) { bv = gm[g]; best = g; }
      gmask |= 1u << best;
    }
    for (int g = 0; g < NGRP; ++g)
      if (!((gmask >> g) & 1u))
        for (int j = 0; j < 8; ++j) row[g * 8 + j] = 0.0f;  // masked gates

    // top-8 experts over masked gates
    int si[TOPK]; float sv[TOPK];
    unsigned long long chosen = 0ull;
    float ssum = 0.0f;
    for (int k = 0; k < TOPK; ++k) {
      int best = 0; float bv = -1.0f;
      for (int e = 0; e < NEXP; ++e)
        if (!((chosen >> e) & 1ull) && row[e] > bv) { bv = row[e]; best = e; }
      chosen |= 1ull << best;
      si[k] = best; sv[k] = bv; ssum += bv;
    }
    ssum = fmaxf(ssum, 1.1920929e-07f);   // clip(.., f32 eps)
    float rinv = 1.0f / ssum;
    int tglob = blk * TPB + threadIdx.x;
    for (int k = 0; k < TOPK; ++k) {
      sel[tglob * TOPK + k] = si[k];
      rp [tglob * TOPK + k] = sv[k] * rinv;
      atomicAdd(&lhist[k][si[k]], 1);
    }
  }
  __syncthreads();

  // ---------------- phase 3: deterministic per-block reductions -----------
  if (threadIdx.x < NEXP) {                 // column sum of masked gates
    float s = 0.0f;
    for (int tl = 0; tl < TPB; ++tl) s += lds[tl][threadIdx.x];
    gsum_blk[blk * NEXP + threadIdx.x] = s;
  }
  for (int i = threadIdx.x; i < TOPK * NEXP; i += blockDim.x)
    hist[blk * (TOPK * NEXP) + i] = ((int*)lhist)[i];
}

// ---------------------------------------------------------------------------
// Kernel 4: per (k,e) exclusive prefix over blocks -> offs; totals -> ktot
// ---------------------------------------------------------------------------
__global__ void hy_scan_blocks_kernel(const int* __restrict__ hist,
                                      int* __restrict__ offs,
                                      int* __restrict__ ktot) {
  int p = threadIdx.x;                       // 0..511 : k = p>>6, e = p&63
  if (p >= TOPK * NEXP) return;
  int k = p >> 6, e = p & 63;
  int run = 0;
  for (int b = 0; b < NBLK; ++b) {
    int idx = (b * TOPK + k) * NEXP + e;
    offs[idx] = run;
    run += hist[idx];
  }
  ktot[p] = run;
}

// ---------------------------------------------------------------------------
// Kernel 5: per-expert prefix over k -> kbase; exp_counts + l_aux
// ---------------------------------------------------------------------------
__global__ void hy_expert_scan_kernel(const int* __restrict__ ktot,
                                      const float* __restrict__ gsum_blk,
                                      int* __restrict__ kbase,
                                      float* __restrict__ out) {
  __shared__ float red[NEXP];
  int e = threadIdx.x;
  if (e < NEXP) {
    int base = 0;
    for (int k = 0; k < TOPK; ++k) {
      kbase[k * NEXP + e] = base;
      base += ktot[k * NEXP + e];
    }
    out[2 + 3 * (size_t)TE + e] = (float)base;   // exp_counts
    float gs = 0.0f;
    for (int b = 0; b < NBLK; ++b) gs += gsum_blk[b * NEXP + e];
    red[e] = (float)base * gs;
  }
  __syncthreads();
  if (e == 0) {
    float s = 0.0f;
    for (int i = 0; i < NEXP; ++i) s += red[i];
    // l_aux = E^2 * mean_e( (cnt/T) * (gsum/T) ) = (E / T^2) * sum
    out[0] = s * ((float)NEXP / ((float)TTOK * (float)TTOK));
  }
}

// ---------------------------------------------------------------------------
// Kernel 6: within-block, within-slot rank via wave32 bit-sliced ballots;
// scatter dense outputs; count valid assignments.
// ---------------------------------------------------------------------------
__global__ __launch_bounds__(128) void hy_rank_kernel(
    const int* __restrict__ sel, const float* __restrict__ rp,
    const int* __restrict__ offs, const int* __restrict__ kbase,
    float* __restrict__ out, int* __restrict__ valid_count) {
  __shared__ int whist[4][NEXP];
  __shared__ int bval;
  const int lane = threadIdx.x & 31;
  const int wave = threadIdx.x >> 5;
  const int blk  = blockIdx.x;
  const int t    = blk * TPB + threadIdx.x;
  if (threadIdx.x == 0) bval = 0;
  float* out_cw = out + 2;
  float* out_tp = out + 2 + (size_t)TE;
  float* out_vm = out + 2 + 2 * (size_t)TE;
  const unsigned lt = (lane == 0) ? 0u : (0xffffffffu >> (32 - lane));

  for (int k = 0; k < TOPK; ++k) {
    for (int i = threadIdx.x; i < 4 * NEXP; i += TPB) ((int*)whist)[i] = 0;
    __syncthreads();
    int   e = sel[t * TOPK + k];
    float p = rp [t * TOPK + k];
    // same-expert lane mask via 6 bit-sliced ballots (E=64 -> 6 bits)
    unsigned same = 0xffffffffu;
#pragma unroll
    for (int b = 0; b < 6; ++b) {
      unsigned bb = ballot32((e >> b) & 1);
      same &= ((e >> b) & 1) ? bb : ~bb;
    }
    int rnk = __builtin_popcount(same & lt);
    if ((same & lt) == 0u) whist[wave][e] = __builtin_popcount(same);
    __syncthreads();
    for (int w = 0; w < wave; ++w) rnk += whist[w][e];

    int pos = kbase[k * NEXP + e] + offs[(blk * TOPK + k) * NEXP + e] + rnk;
    bool valid = pos < CAPACITY;
    if (valid) {
      size_t o = (size_t)t * NEXP + e;
      out_cw[o] = p;             // combine_weights
      out_tp[o] = (float)pos;    // token_priority
      out_vm[o] = 1.0f;          // valid_mask
    }
    unsigned vb = ballot32(valid);
    if (lane == 0) atomicAdd(&bval, __builtin_popcount(vb));
    __syncthreads();
  }
  if (threadIdx.x == 0) atomicAdd(valid_count, bval);
}

// ---------------------------------------------------------------------------
// Kernel 7: exp_capacity_rate
// ---------------------------------------------------------------------------
__global__ void hy_rate_kernel(const int* __restrict__ valid_count,
                               float* __restrict__ out) {
  if (threadIdx.x == 0)
    out[1] = (float)(*valid_count) / (float)(TTOK * TOPK);
}

// ---------------------------------------------------------------------------
extern "C" void kernel_launch(void* const* d_in, const int* in_sizes, int n_in,
                              void* d_out, int out_size, void* d_ws, size_t ws_size,
                              hipStream_t stream) {
  const float* X = (const float*)d_in[0];   // hidden_states [4,4096,4096] f32
  const float* W = (const float*)d_in[1];   // wg_weight     [64,4096]     f32
  float* out = (float*)d_out;

  char* ws = (char*)d_ws;
  int*    sel      = (int*)   (ws + 0);                           // 512 KB
  float*  rp       = (float*) (ws + (1u << 19));                  // 512 KB
  int*    hist     = (int*)   (ws + (2u << 19));                  // 256 KB
  int*    offs     = (int*)   (ws + (2u << 19) + (1u << 18));     // 256 KB
  __bf16* Wb       = (__bf16*)(ws + (3u << 19));                  // 512 KB
  float*  gsum_blk = (float*) (ws + (4u << 19));                  //  32 KB
  int*    ktot     = (int*)   (ws + (4u << 19) + (1u << 15));     //   2 KB
  int*    kbase    = (int*)   (ws + (4u << 19) + (1u << 15) + 2048);
  int*    valid_ct = (int*)   (ws + (4u << 19) + (1u << 16));

  hy_zero_kernel<<<2048, 512, 0, stream>>>(out, out_size, valid_ct);
  hy_prep_w_kernel<<<(4 * KITERS * 32 * 16) / 256, 256, 0, stream>>>(W, Wb);
  hy_gate_kernel<<<NBLK, 256, 0, stream>>>(X, Wb, sel, rp, hist, gsum_blk);
  hy_scan_blocks_kernel<<<1, 512, 0, stream>>>(hist, offs, ktot);
  hy_expert_scan_kernel<<<1, 64, 0, stream>>>(ktot, gsum_blk, kbase, out);
  hy_rank_kernel<<<NBLK, TPB, 0, stream>>>(sel, rp, offs, kbase, out, valid_ct);
  hy_rate_kernel<<<1, 32, 0, stream>>>(valid_ct, out);
}
